// MultiHeadAttention_64003602645570
// MI455X (gfx1250) — matrix operand order repaired and checked by reading, not compiled, not run on hardware
//
#include <hip/hip_runtime.h>

// ---------------- constants ----------------
#define BZ 2
#define SQ 2048
#define EM 1024
#define NH 16
#define DH 64
#define SCALE 0.125f   // 1/sqrt(64)

typedef __attribute__((ext_vector_type(16))) __bf16 v16bf;
typedef __attribute__((ext_vector_type(8)))  float  v8f;
typedef __attribute__((ext_vector_type(4)))  unsigned int u32x4;

union FragU { u32x4 u[2]; v16bf v; };

__device__ __forceinline__ unsigned short f2bf(float x) {
    unsigned int b = __float_as_uint(x);
    b += 0x7FFFu + ((b >> 16) & 1u);          // round-to-nearest-even
    return (unsigned short)(b >> 16);
}

// A-fragment: two 16B chunks (elements 0-7 and 8-15)
__device__ __forceinline__ v16bf ldA(const unsigned short* p0, const unsigned short* p1) {
    FragU f;
    f.u[0] = *(const u32x4*)p0;
    f.u[1] = *(const u32x4*)p1;
    return f.v;
}
// B-fragment: one contiguous 32B chunk
__device__ __forceinline__ v16bf ldB(const unsigned short* p) {
    FragU f;
    f.u[0] = *(const u32x4*)p;
    f.u[1] = *(const u32x4*)(p + 8);
    return f.v;
}

__device__ __forceinline__ v8f wmma_bf16(v16bf a, v16bf b, v8f c) {
    return __builtin_amdgcn_wmma_f32_16x16x32_bf16(false, a, false, b, (short)0, c, false, false);
}

// ---------------- f32 -> bf16 convert ----------------
__global__ void k_cvt_bf16(const float* __restrict__ src, unsigned short* __restrict__ dst, int n) {
    int i = blockIdx.x * 256 + threadIdx.x;
    if (i < n) dst[i] = f2bf(src[i]);
}

// ---------------- repack W_{q,k,v} into B-fragment order ----------------
// layout: [which][h][ntile(4)][kstep(32)][lane(32)][elem(16)]
__global__ void k_repack_w(const float* __restrict__ wq, const float* __restrict__ wk,
                           const float* __restrict__ wv, unsigned short* __restrict__ wpack) {
    int p = blockIdx.x * 256 + threadIdx.x;   // 3 * 2^20 total
    int i  = p & 15;
    int L  = (p >> 4) & 31;
    int ks = (p >> 9) & 31;
    int nt = (p >> 14) & 3;
    int h  = (p >> 16) & 15;
    int which = p >> 20;
    const float* W = (which == 0) ? wq : ((which == 1) ? wk : wv);
    int kg = ks * 32 + i + 16 * (L >> 4);     // B-matrix k index
    int n  = nt * 16 + (L & 15);              // B-matrix n index
    wpack[p] = f2bf(W[(size_t)h * (EM * DH) + (size_t)kg * DH + n]);
}

// ---------------- QKV projection: 32 rows x 64 cols per wave, pipelined ----------------
__global__ void __launch_bounds__(128) k_qkv(const unsigned short* __restrict__ xb,
                                             const unsigned short* __restrict__ wpack,
                                             unsigned short* __restrict__ qb,
                                             unsigned short* __restrict__ kbuf,
                                             unsigned short* __restrict__ vtb) {
    const int lane = threadIdx.x & 31;
    const int gw   = (blockIdx.x * 128 + threadIdx.x) >> 5;  // 0..6143
    const int which = gw >> 11;
    const int rem   = gw & 2047;
    const int b  = rem >> 10;
    const int h  = (rem >> 6) & 15;
    const int mt = rem & 63;                 // 64 tiles of 32 rows
    const int lm = lane & 15, half = lane >> 4;

    const unsigned short* xrow0 = xb + (size_t)(b * SQ + mt * 32 + lm) * EM;
    const unsigned short* xrow1 = xrow0 + (size_t)16 * EM;
    const unsigned short* wp    = wpack + (size_t)((which * 16 + h) * 4) * (32 * 32 * 16);

    v8f z = {0.f,0.f,0.f,0.f,0.f,0.f,0.f,0.f};
    v8f acc0[4] = {z, z, z, z};
    v8f acc1[4] = {z, z, z, z};

    // preload k-step 0
    v16bf a0c = ldA(xrow0 + half * 16, xrow0 + 8 + half * 16);
    v16bf a1c = ldA(xrow1 + half * 16, xrow1 + 8 + half * 16);
    v16bf bc[4];
#pragma unroll
    for (int j = 0; j < 4; ++j) bc[j] = ldB(wp + (size_t)((j * 32) * 32 + lane) * 16);

#pragma unroll 2
    for (int ks = 0; ks < 31; ++ks) {
        const int k1 = (ks + 1) * 32;
        // issue next k-step loads before consuming current fragments
        v16bf a0n = ldA(xrow0 + k1 + half * 16, xrow0 + k1 + 8 + half * 16);
        v16bf a1n = ldA(xrow1 + k1 + half * 16, xrow1 + k1 + 8 + half * 16);
        v16bf bn[4];
#pragma unroll
        for (int j = 0; j < 4; ++j) bn[j] = ldB(wp + (size_t)((j * 32 + ks + 1) * 32 + lane) * 16);
#pragma unroll
        for (int j = 0; j < 4; ++j) acc0[j] = wmma_bf16(a0c, bc[j], acc0[j]);
#pragma unroll
        for (int j = 0; j < 4; ++j) acc1[j] = wmma_bf16(a1c, bc[j], acc1[j]);
        a0c = a0n; a1c = a1n;
#pragma unroll
        for (int j = 0; j < 4; ++j) bc[j] = bn[j];
    }
#pragma unroll
    for (int j = 0; j < 4; ++j) acc0[j] = wmma_bf16(a0c, bc[j], acc0[j]);
#pragma unroll
    for (int j = 0; j < 4; ++j) acc1[j] = wmma_bf16(a1c, bc[j], acc1[j]);

    const size_t bh = (size_t)(b * NH + h);
#pragma unroll
    for (int t = 0; t < 2; ++t) {
        v8f* acc = t ? acc1 : acc0;
#pragma unroll
        for (int j = 0; j < 4; ++j) {
#pragma unroll
            for (int r = 0; r < 8; ++r) {
                int srow = mt * 32 + t * 16 + r + 8 * half;
                int d    = j * 16 + lm;
                unsigned short v = f2bf(acc[j][r]);
                if (which == 0)      qb  [(bh * SQ + srow) * DH + d] = v;
                else if (which == 1) kbuf[(bh * SQ + srow) * DH + d] = v;
                else                 vtb [(bh * DH + d) * SQ + srow] = v;  // transposed
            }
        }
    }
}

// ---------------- causal flash attention, one (b,h,16-row q-tile) per wave ----------------
__global__ void __launch_bounds__(128) k_flash(const unsigned short* __restrict__ qb,
                                               const unsigned short* __restrict__ kbuf,
                                               const unsigned short* __restrict__ vtb,
                                               unsigned short* __restrict__ attnb) {
    const int w    = threadIdx.x >> 5;          // wave in block
    const int lane = threadIdx.x & 31;
    const int gw   = blockIdx.x * 4 + w;        // 0..4095
    const int b  = gw >> 11;
    const int h  = (gw >> 7) & 15;
    const int qt = gw & 127;
    const int lm = lane & 15, half = lane >> 4;

    __shared__ __align__(16) unsigned short pb[4][16 * 32];
    unsigned short* myp = pb[w];

    const size_t bh = (size_t)(b * NH + h);
    const unsigned short* qrow  = qb   + (bh * SQ + qt * 16 + lm) * DH;
    const unsigned short* kbase = kbuf + bh * SQ * DH;
    const unsigned short* vbase = vtb  + bh * DH * SQ;

    // Q fragments held for whole loop (K-dim = 64 -> two 16x32 A frags)
    v16bf qf0 = ldA(qrow +      half * 16, qrow +  8 + half * 16);
    v16bf qf1 = ldA(qrow + 32 + half * 16, qrow + 40 + half * 16);

    v8f z = {0.f,0.f,0.f,0.f,0.f,0.f,0.f,0.f};
    v8f acc[4] = {z, z, z, z};
    float m_r[8], l_r[8];
#pragma unroll
    for (int r = 0; r < 8; ++r) { m_r[r] = -3.0e38f; l_r[r] = 0.f; }

    const int nkb = ((qt * 16 + 15) >> 5) + 1;   // causal block bound

    // preload K fragments for block 0
    v16bf kf[4];
    {
        const unsigned short* kr0 = kbase + (size_t)lm * DH;
        const unsigned short* kr1 = kr0 + 16 * DH;
        kf[0] = ldB(kr0 +      half * 16);
        kf[1] = ldB(kr0 + 32 + half * 16);
        kf[2] = ldB(kr1 +      half * 16);
        kf[3] = ldB(kr1 + 32 + half * 16);
    }

    for (int kb = 0; kb < nkb; ++kb) {
        // issue V loads for this block early (consumed only after softmax)
        v16bf vf[4];
#pragma unroll
        for (int j = 0; j < 4; ++j)
            vf[j] = ldB(vbase + (size_t)(j * 16 + lm) * SQ + kb * 32 + half * 16);

        // ---- scores: two 16x16 C tiles covering 32 k-positions ----
        v8f c0 = z, c1 = z;
        c0 = wmma_bf16(qf0, kf[0], c0); c0 = wmma_bf16(qf1, kf[1], c0);
        c1 = wmma_bf16(qf0, kf[2], c1); c1 = wmma_bf16(qf1, kf[3], c1);

        // prefetch next block's K fragments (clamped; hidden under softmax)
        const int kbn = (kb + 1 < nkb) ? kb + 1 : kb;
        v16bf kn[4];
        {
            const unsigned short* kr0 = kbase + (size_t)(kbn * 32 + lm) * DH;
            const unsigned short* kr1 = kr0 + 16 * DH;
            kn[0] = ldB(kr0 +      half * 16);
            kn[1] = ldB(kr0 + 32 + half * 16);
            kn[2] = ldB(kr1 +      half * 16);
            kn[3] = ldB(kr1 + 32 + half * 16);
        }

        // ---- online softmax (f32), per C-register row stats ----
#pragma unroll
        for (int r = 0; r < 8; ++r) {
            int qg  = qt * 16 + r + 8 * half;
            int kg0 = kb * 32 + lm;
            float v0 = c0[r] * SCALE;
            float v1 = c1[r] * SCALE;
            if (kg0 > qg)      v0 = -1.0e30f;
            if (kg0 + 16 > qg) v1 = -1.0e30f;
            float rm = fmaxf(v0, v1);
            rm = fmaxf(rm, __shfl_xor(rm, 1));
            rm = fmaxf(rm, __shfl_xor(rm, 2));
            rm = fmaxf(rm, __shfl_xor(rm, 4));
            rm = fmaxf(rm, __shfl_xor(rm, 8));
            float mn = fmaxf(m_r[r], rm);
            float alpha = __expf(m_r[r] - mn);
            float p0 = __expf(v0 - mn);
            float p1 = __expf(v1 - mn);
            float rs = p0 + p1;
            rs += __shfl_xor(rs, 1);
            rs += __shfl_xor(rs, 2);
            rs += __shfl_xor(rs, 4);
            rs += __shfl_xor(rs, 8);
            l_r[r] = l_r[r] * alpha + rs;
            m_r[r] = mn;
            acc[0][r] *= alpha; acc[1][r] *= alpha;
            acc[2][r] *= alpha; acc[3][r] *= alpha;
            int mrow = r + 8 * half;
            myp[mrow * 32 +      lm] = f2bf(p0);
            myp[mrow * 32 + 16 + lm] = f2bf(p1);
        }

        // ---- P (C-layout) -> A-layout via private LDS tile; then P x V ----
        const unsigned short* pr = myp + lm * 32;
        v16bf ap = ldA(pr + half * 16, pr + 8 + half * 16);
#pragma unroll
        for (int j = 0; j < 4; ++j) acc[j] = wmma_bf16(ap, vf[j], acc[j]);

#pragma unroll
        for (int j = 0; j < 4; ++j) kf[j] = kn[j];
    }

    // ---- epilogue: normalize, store bf16 [B, S, H*D] ----
#pragma unroll
    for (int j = 0; j < 4; ++j) {
#pragma unroll
        for (int r = 0; r < 8; ++r) {
            float o = acc[j][r] / l_r[r];
            int srow = qt * 16 + r + 8 * half;
            attnb[((size_t)b * SQ + srow) * EM + h * DH + j * 16 + lm] = f2bf(o);
        }
    }
}

// ---------------- output projection: 32 rows x 64 cols per wave, pipelined ----------------
__global__ void __launch_bounds__(128) k_proj(const unsigned short* __restrict__ attnb,
                                              const unsigned short* __restrict__ wpb,
                                              const float* __restrict__ bias,
                                              float* __restrict__ out) {
    const int lane = threadIdx.x & 31;
    const int gw   = (blockIdx.x * 128 + threadIdx.x) >> 5;  // 0..2047
    const int mt = gw >> 4;     // 128 tiles of 32 rows
    const int ng = gw & 15;     // 16 groups of 64 cols
    const int lm = lane & 15, half = lane >> 4;

    const unsigned short* arow0 = attnb + (size_t)(mt * 32 + lm) * EM;
    const unsigned short* arow1 = arow0 + (size_t)16 * EM;
    const unsigned short* wcol  = wpb + (size_t)(ng * 64 + lm) * EM + half * 16;

    v8f z = {0.f,0.f,0.f,0.f,0.f,0.f,0.f,0.f};
    v8f acc0[4] = {z, z, z, z};
    v8f acc1[4] = {z, z, z, z};

    v16bf a0c = ldA(arow0 + half * 16, arow0 + 8 + half * 16);
    v16bf a1c = ldA(arow1 + half * 16, arow1 + 8 + half * 16);
    v16bf bc[4];
#pragma unroll
    for (int j = 0; j < 4; ++j) bc[j] = ldB(wcol + (size_t)(j * 16) * EM);

#pragma unroll 2
    for (int ks = 0; ks < 31; ++ks) {
        const int k1 = (ks + 1) * 32;
        v16bf a0n = ldA(arow0 + k1 + half * 16, arow0 + k1 + 8 + half * 16);
        v16bf a1n = ldA(arow1 + k1 + half * 16, arow1 + k1 + 8 + half * 16);
        v16bf bn[4];
#pragma unroll
        for (int j = 0; j < 4; ++j) bn[j] = ldB(wcol + (size_t)(j * 16) * EM + k1);
#pragma unroll
        for (int j = 0; j < 4; ++j) acc0[j] = wmma_bf16(a0c, bc[j], acc0[j]);
#pragma unroll
        for (int j = 0; j < 4; ++j) acc1[j] = wmma_bf16(a1c, bc[j], acc1[j]);
        a0c = a0n; a1c = a1n;
#pragma unroll
        for (int j = 0; j < 4; ++j) bc[j] = bn[j];
    }
#pragma unroll
    for (int j = 0; j < 4; ++j) acc0[j] = wmma_bf16(a0c, bc[j], acc0[j]);
#pragma unroll
    for (int j = 0; j < 4; ++j) acc1[j] = wmma_bf16(a1c, bc[j], acc1[j]);

#pragma unroll
    for (int t = 0; t < 2; ++t) {
        v8f* acc = t ? acc1 : acc0;
#pragma unroll
        for (int j = 0; j < 4; ++j) {
#pragma unroll
            for (int r = 0; r < 8; ++r) {
                int row = mt * 32 + t * 16 + r + 8 * half;
                int col = ng * 64 + j * 16 + lm;
                out[(size_t)row * EM + col] = acc[j][r] + bias[col];
            }
        }
    }
}

// ---------------- launch ----------------
extern "C" void kernel_launch(void* const* d_in, const int* in_sizes, int n_in,
                              void* d_out, int out_size, void* d_ws, size_t ws_size,
                              hipStream_t stream) {
    const float* x      = (const float*)d_in[0];
    const float* wq     = (const float*)d_in[1];
    const float* wk     = (const float*)d_in[2];
    const float* wv     = (const float*)d_in[3];
    const float* w_proj = (const float*)d_in[4];
    const float* b_proj = (const float*)d_in[5];
    float* out = (float*)d_out;

    char* ws = (char*)d_ws;
    size_t off = 0;
    unsigned short* xb    = (unsigned short*)(ws + off); off += (size_t)BZ * SQ * EM * 2;        // 8 MB
    unsigned short* wpack = (unsigned short*)(ws + off); off += (size_t)3 * NH * EM * DH * 2;    // 6 MB
    unsigned short* qb    = (unsigned short*)(ws + off); off += (size_t)BZ * NH * SQ * DH * 2;   // 8 MB
    unsigned short* kbuf  = (unsigned short*)(ws + off); off += (size_t)BZ * NH * SQ * DH * 2;   // 8 MB
    unsigned short* vtb   = (unsigned short*)(ws + off); off += (size_t)BZ * NH * DH * SQ * 2;   // 8 MB
    unsigned short* attnb = (unsigned short*)(ws + off); off += (size_t)BZ * SQ * EM * 2;        // 8 MB
    unsigned short* wpb   = (unsigned short*)(ws + off); off += (size_t)EM * EM * 2;             // 2 MB
    (void)ws_size; (void)in_sizes; (void)n_in; (void)out_size;

    // 1) bf16 conversions
    k_cvt_bf16<<<(BZ * SQ * EM + 255) / 256, 256, 0, stream>>>(x, xb, BZ * SQ * EM);
    k_cvt_bf16<<<(EM * EM + 255) / 256, 256, 0, stream>>>(w_proj, wpb, EM * EM);
    // 2) repack QKV weights into WMMA B-fragment order
    k_repack_w<<<(3 * NH * EM * DH + 255) / 256, 256, 0, stream>>>(wq, wk, wv, wpack);
    // 3) QKV projections (6144 waves, 32x64 tile per wave)
    k_qkv<<<1536, 128, 0, stream>>>(xb, wpack, qb, kbuf, vtb);
    // 4) causal flash attention (4096 waves)
    k_flash<<<1024, 128, 0, stream>>>(qb, kbuf, vtb, attnb);
    // 5) output projection + bias (2048 waves, 32x64 tile per wave)
    k_proj<<<512, 128, 0, stream>>>(attnb, wpb, b_proj, out);
}
